// TriangleMultiplicationOutgoing_86019605004402
// MI455X (gfx1250) — compile-verified, matrix operand order checked by reading, not run on hardware
//
#include <hip/hip_runtime.h>
#include <hip/hip_bf16.h>

#define L   512
#define CZ  128
#define LL  (L * L)
#define LN_EPS 1e-5f

typedef _Float16 f16;
typedef __attribute__((ext_vector_type(16))) _Float16 v16h;
typedef __attribute__((ext_vector_type(8)))  _Float16 v8h;
typedef __attribute__((ext_vector_type(8)))  float    v8f;
typedef __attribute__((ext_vector_type(4)))  float    v4f;

// ---------------------------------------------------------------------------
// WMMA wrapper: D = A(16x32 f16) * B(32x16 f16) + C(16x16 f32)
// ---------------------------------------------------------------------------
__device__ __forceinline__ v8f wmma_fp16(v16h a, v16h b, v8f c) {
  // (neg_a, A, neg_b, B, c_mod, C, reuse_a, reuse_b)
  return __builtin_amdgcn_wmma_f32_16x16x32_f16(false, a, false, b, (short)0, c,
                                                false, false);
}

// ---------------------------------------------------------------------------
// Load a 16x32 f16 fragment (A layout; B layout is the mirrored N<->M form,
// served by storing B-sources as [n][k]).  Per ISA 7.12.2 (16-bit A 16x32):
//   lane l:  row = rbase + (l & 15),  K-halfbase = 8*(l>>4)
//   VGPR 0..3 -> K = halfbase + 0..7   (contiguous -> one 16B load)
//   VGPR 4..7 -> K = halfbase + 16..23 (contiguous -> one 16B load)
// ---------------------------------------------------------------------------
__device__ __forceinline__ v16h load_frag(const f16* base, int rbase, int stride,
                                          int k0, int lane) {
  const f16* p = base + (size_t)(rbase + (lane & 15)) * stride + k0 + 8 * (lane >> 4);
  v8h lo = *(const v8h*)(p);
  v8h hi = *(const v8h*)(p + 16);
  v16h f;
#pragma unroll
  for (int i = 0; i < 8; ++i) { f[i] = lo[i]; f[i + 8] = hi[i]; }
  return f;
}

// ---------------------------------------------------------------------------
// Async-copy 32 rows x 64 bytes (a 32x32 f16 K-chunk) from global to LDS.
// Lane l copies row l; 4 x GLOBAL_LOAD_ASYNC_TO_LDS_B128 (ASYNCcnt-tracked,
// no VGPR landing zone).  LDS row stride = 80B (16B aligned, bank-friendly).
// ---------------------------------------------------------------------------
__device__ __forceinline__ void async_copy_chunk(f16* lds_dst, const f16* gsrc,
                                                 int lane) {
  unsigned lo = (unsigned)(uintptr_t)lds_dst + (unsigned)lane * 80u;
  uint64_t ga = (uint64_t)(uintptr_t)(gsrc + (size_t)lane * L);
#pragma unroll
  for (int t = 0; t < 4; ++t) {
    unsigned l = lo + t * 16u;
    uint64_t g = ga + (uint64_t)(t * 16);
    asm volatile("global_load_async_to_lds_b128 %0, %1, off"
                 :: "v"(l), "v"(g) : "memory");
  }
}

__device__ __forceinline__ float sigmoidf_(float x) {
  return 1.0f / (1.0f + __expf(-x));
}

__device__ __forceinline__ float wave_sum32(float v) {
#pragma unroll
  for (int o = 16; o > 0; o >>= 1) v += __shfl_xor(v, o, 32);
  return v;
}

// ---------------------------------------------------------------------------
// Kernel 1: convert six 128x128 f32 weights to f16, transposed to [n][k]
// slot order: 0=gate_a 1=proj_a 2=gate_b 3=proj_b 4=gate_out 5=proj_out
// ---------------------------------------------------------------------------
__global__ void k_wconv(const float* __restrict__ Wga, const float* __restrict__ Wpa,
                        const float* __restrict__ Wgb, const float* __restrict__ Wpb,
                        const float* __restrict__ Wgo, const float* __restrict__ Wpo,
                        f16* __restrict__ wt) {
  const float* src[6] = {Wga, Wpa, Wgb, Wpb, Wgo, Wpo};
  const float* w = src[blockIdx.x];
  f16* dst = wt + (size_t)blockIdx.x * (CZ * CZ);
  for (int idx = threadIdx.x; idx < CZ * CZ; idx += blockDim.x) {
    int k = idx >> 7, n = idx & 127;     // w is [k][n] row-major
    dst[n * CZ + k] = (f16)w[idx];
  }
}

// ---------------------------------------------------------------------------
// Kernel 2: fused input LayerNorm + 5 projections + sigmoid gating + mask.
// Block = 256 threads (8 wave32), 128 flattened rows per block.
// a_ws/b_ws stored channel-major [c][row] (f16); g_ws row-major [row][c].
// ---------------------------------------------------------------------------
__global__ __launch_bounds__(256) void k_proj(
    const float* __restrict__ z, const int* __restrict__ mask,
    const float* __restrict__ lnw, const float* __restrict__ lnb,
    const f16* __restrict__ wt,
    f16* __restrict__ a_ws, f16* __restrict__ b_ws, f16* __restrict__ g_ws) {
  __shared__ f16 zln[128 * 136];          // stride 136 f16 = 272B: conflict-free
  const int stride = 136;
  const int lane = threadIdx.x & 31;
  const int wave = threadIdx.x >> 5;
  const int gr0  = blockIdx.x * 128 + wave * 16;   // this wave's 16 rows

  // ---- LayerNorm: 16 rows per wave, 4 channels per lane ----
  v4f w4 = *(const v4f*)(lnw + lane * 4);
  v4f b4 = *(const v4f*)(lnb + lane * 4);
  for (int r = 0; r < 16; ++r) {
    v4f x = *(const v4f*)(z + (size_t)(gr0 + r) * CZ + lane * 4);
    float s  = x.x + x.y + x.z + x.w;
    float sq = x.x * x.x + x.y * x.y + x.z * x.z + x.w * x.w;
    s  = wave_sum32(s);
    sq = wave_sum32(sq);
    float mu   = s * (1.0f / CZ);
    float var  = sq * (1.0f / CZ) - mu * mu;
    float rstd = rsqrtf(var + LN_EPS);
    f16* q = &zln[(wave * 16 + r) * stride + lane * 4];
    q[0] = (f16)((x.x - mu) * rstd * w4.x + b4.x);
    q[1] = (f16)((x.y - mu) * rstd * w4.y + b4.y);
    q[2] = (f16)((x.z - mu) * rstd * w4.z + b4.z);
    q[3] = (f16)((x.w - mu) * rstd * w4.w + b4.w);
  }
  __syncthreads();

  // ---- per-row mask in C-tile layout: VGPR r <-> row 8*(lane>>4)+r ----
  const int moff = (lane >> 4) * 8;
  float mk[8];
#pragma unroll
  for (int r = 0; r < 8; ++r) mk[r] = (float)mask[gr0 + moff + r];

  // ---- preload this wave's A fragments (16 rows x full K=128) ----
  v16h A[4];
#pragma unroll
  for (int ks = 0; ks < 4; ++ks) A[ks] = load_frag(zln, wave * 16, stride, ks * 32, lane);

  const f16* Wga = wt;
  const f16* Wpa = wt + 16384;
  const f16* Wgb = wt + 2 * 16384;
  const f16* Wpb = wt + 3 * 16384;
  const f16* Wgo = wt + 4 * 16384;

  for (int nt = 0; nt < 8; ++nt) {
    const int n0 = nt * 16;
    const int n  = n0 + (lane & 15);

    // a = sigmoid(zln@Wga) * (zln@Wpa) * mask
    v8f accG = {0, 0, 0, 0, 0, 0, 0, 0};
    v8f accP = {0, 0, 0, 0, 0, 0, 0, 0};
#pragma unroll
    for (int ks = 0; ks < 4; ++ks) {
      v16h Bg = load_frag(Wga, n0, CZ, ks * 32, lane);
      v16h Bp = load_frag(Wpa, n0, CZ, ks * 32, lane);
      accG = wmma_fp16(A[ks], Bg, accG);
      accP = wmma_fp16(A[ks], Bp, accP);
    }
    v8h pa;
#pragma unroll
    for (int r = 0; r < 8; ++r)
      pa[r] = (f16)(sigmoidf_(accG[r]) * accP[r] * mk[r]);
    *(v8h*)(a_ws + (size_t)n * LL + gr0 + moff) = pa;   // [c][row], 16B store

    // b = sigmoid(zln@Wgb) * (zln@Wpb) * mask
    v8f bG = {0, 0, 0, 0, 0, 0, 0, 0};
    v8f bP = {0, 0, 0, 0, 0, 0, 0, 0};
#pragma unroll
    for (int ks = 0; ks < 4; ++ks) {
      v16h Bg = load_frag(Wgb, n0, CZ, ks * 32, lane);
      v16h Bp = load_frag(Wpb, n0, CZ, ks * 32, lane);
      bG = wmma_fp16(A[ks], Bg, bG);
      bP = wmma_fp16(A[ks], Bp, bP);
    }
    v8h pb;
#pragma unroll
    for (int r = 0; r < 8; ++r)
      pb[r] = (f16)(sigmoidf_(bG[r]) * bP[r] * mk[r]);
    *(v8h*)(b_ws + (size_t)n * LL + gr0 + moff) = pb;

    // g = sigmoid(zln@Wgo)   (mask applied at the end, per reference)
    v8f gA = {0, 0, 0, 0, 0, 0, 0, 0};
#pragma unroll
    for (int ks = 0; ks < 4; ++ks) {
      v16h Bg = load_frag(Wgo, n0, CZ, ks * 32, lane);
      gA = wmma_fp16(A[ks], Bg, gA);
    }
#pragma unroll
    for (int r = 0; r < 8; ++r)
      g_ws[(size_t)(gr0 + moff + r) * CZ + n] = (f16)sigmoidf_(gA[r]);
  }
}

// ---------------------------------------------------------------------------
// Kernel 3: m[i,j,c] = sum_k a[i,k,c]*b[j,k,c].  One channel per wave,
// 32x32 (i,j) tile per wave, K=512 streamed as 16 WMMA steps with a
// double-buffered GLOBAL_LOAD_ASYNC_TO_LDS pipeline (ASYNCcnt-tracked).
// a_ws/b_ws are [c][row][k]; both operands use the same fragment loader.
// grid = (16, 16, 32); block = 128 (4 waves = 4 channels, 10KB LDS/wave).
// ---------------------------------------------------------------------------
#define CH_STRIDE 40                     // padded f16 row stride in LDS (80B)
#define CH_ELEMS  (32 * CH_STRIDE)      // one 32x32 chunk buffer

__global__ __launch_bounds__(128) void k_einsum(
    const f16* __restrict__ a_ws, const f16* __restrict__ b_ws,
    float* __restrict__ m_ws) {
  // [wave][buf][A/B][32 x 40 f16]  = 4*2*2*2560B = 40KB
  __shared__ __align__(16) f16 stage[4 * 2 * 2 * CH_ELEMS];
  const int lane = threadIdx.x & 31;
  const int wave = threadIdx.x >> 5;
  const int c  = blockIdx.z * 4 + wave;
  const int i0 = blockIdx.x * 32;
  const int j0 = blockIdx.y * 32;
  const f16* Ac = a_ws + (size_t)c * LL + (size_t)i0 * L;
  const f16* Bc = b_ws + (size_t)c * LL + (size_t)j0 * L;
  f16* bufA[2] = {&stage[((wave * 2 + 0) * 2 + 0) * CH_ELEMS],
                  &stage[((wave * 2 + 1) * 2 + 0) * CH_ELEMS]};
  f16* bufB[2] = {&stage[((wave * 2 + 0) * 2 + 1) * CH_ELEMS],
                  &stage[((wave * 2 + 1) * 2 + 1) * CH_ELEMS]};

  // prologue: stage k-chunk 0 into buffer 0 (8 async b128 ops)
  async_copy_chunk(bufA[0], Ac, lane);
  async_copy_chunk(bufB[0], Bc, lane);

  v8f acc[2][2] = {};
  for (int ks = 0; ks < 16; ++ks) {
    const int cur = ks & 1;
    if (ks < 15) {
      // stage next chunk while computing on current one
      async_copy_chunk(bufA[cur ^ 1], Ac + (ks + 1) * 32, lane);
      async_copy_chunk(bufB[cur ^ 1], Bc + (ks + 1) * 32, lane);
      // async loads complete in order: <=8 outstanding => current chunk done
      asm volatile("s_wait_asynccnt 8" ::: "memory");
    } else {
      asm volatile("s_wait_asynccnt 0" ::: "memory");
    }
    v16h A0 = load_frag(bufA[cur], 0,  CH_STRIDE, 0, lane);
    v16h A1 = load_frag(bufA[cur], 16, CH_STRIDE, 0, lane);
    v16h B0 = load_frag(bufB[cur], 0,  CH_STRIDE, 0, lane);
    v16h B1 = load_frag(bufB[cur], 16, CH_STRIDE, 0, lane);
    acc[0][0] = wmma_fp16(A0, B0, acc[0][0]);
    acc[0][1] = wmma_fp16(A0, B1, acc[0][1]);
    acc[1][0] = wmma_fp16(A1, B0, acc[1][0]);
    acc[1][1] = wmma_fp16(A1, B1, acc[1][1]);
  }

  const int moff = (lane >> 4) * 8;
  const int n    = lane & 15;
#pragma unroll
  for (int it = 0; it < 2; ++it)
#pragma unroll
    for (int jt = 0; jt < 2; ++jt)
#pragma unroll
      for (int r = 0; r < 8; ++r) {
        int gi = i0 + it * 16 + moff + r;
        int gj = j0 + jt * 16 + n;
        m_ws[(size_t)(gi * L + gj) * CZ + c] = acc[it][jt][r];
      }
}

// ---------------------------------------------------------------------------
// Kernel 4: dz = g * LayerNorm(m) @ W_proj_out * mask
// ---------------------------------------------------------------------------
__global__ __launch_bounds__(256) void k_final(
    const float* __restrict__ m_ws, const int* __restrict__ mask,
    const float* __restrict__ lnw, const float* __restrict__ lnb,
    const f16* __restrict__ Wpo, const f16* __restrict__ g_ws,
    float* __restrict__ out) {
  __shared__ f16 mln[128 * 136];
  const int stride = 136;
  const int lane = threadIdx.x & 31;
  const int wave = threadIdx.x >> 5;
  const int gr0  = blockIdx.x * 128 + wave * 16;

  v4f w4 = *(const v4f*)(lnw + lane * 4);
  v4f b4 = *(const v4f*)(lnb + lane * 4);
  for (int r = 0; r < 16; ++r) {
    v4f x = *(const v4f*)(m_ws + (size_t)(gr0 + r) * CZ + lane * 4);
    float s  = x.x + x.y + x.z + x.w;
    float sq = x.x * x.x + x.y * x.y + x.z * x.z + x.w * x.w;
    s  = wave_sum32(s);
    sq = wave_sum32(sq);
    float mu   = s * (1.0f / CZ);
    float var  = sq * (1.0f / CZ) - mu * mu;
    float rstd = rsqrtf(var + LN_EPS);
    f16* q = &mln[(wave * 16 + r) * stride + lane * 4];
    q[0] = (f16)((x.x - mu) * rstd * w4.x + b4.x);
    q[1] = (f16)((x.y - mu) * rstd * w4.y + b4.y);
    q[2] = (f16)((x.z - mu) * rstd * w4.z + b4.z);
    q[3] = (f16)((x.w - mu) * rstd * w4.w + b4.w);
  }
  __syncthreads();

  const int moff = (lane >> 4) * 8;
  float mk[8];
#pragma unroll
  for (int r = 0; r < 8; ++r) mk[r] = (float)mask[gr0 + moff + r];

  v16h A[4];
#pragma unroll
  for (int ks = 0; ks < 4; ++ks) A[ks] = load_frag(mln, wave * 16, stride, ks * 32, lane);

  for (int nt = 0; nt < 8; ++nt) {
    const int n0 = nt * 16;
    const int n  = n0 + (lane & 15);
    v8f acc = {0, 0, 0, 0, 0, 0, 0, 0};
#pragma unroll
    for (int ks = 0; ks < 4; ++ks) {
      v16h B = load_frag(Wpo, n0, CZ, ks * 32, lane);
      acc = wmma_fp16(A[ks], B, acc);
    }
#pragma unroll
    for (int r = 0; r < 8; ++r) {
      size_t rf = (size_t)(gr0 + moff + r);
      float g = (float)g_ws[rf * CZ + n];
      out[rf * CZ + n] = g * acc[r] * mk[r];
    }
  }
}

// ---------------------------------------------------------------------------
// Host launcher
// ---------------------------------------------------------------------------
extern "C" void kernel_launch(void* const* d_in, const int* in_sizes, int n_in,
                              void* d_out, int out_size, void* d_ws, size_t ws_size,
                              hipStream_t stream) {
  (void)in_sizes; (void)n_in; (void)out_size; (void)ws_size;
  const float* z      = (const float*)d_in[0];
  const int*   mask   = (const int*)d_in[1];
  const float* lnin_w = (const float*)d_in[2];
  const float* lnin_b = (const float*)d_in[3];
  const float* lnout_w= (const float*)d_in[4];
  const float* lnout_b= (const float*)d_in[5];
  const float* Wpa    = (const float*)d_in[6];
  const float* Wga    = (const float*)d_in[7];
  const float* Wpb    = (const float*)d_in[8];
  const float* Wgb    = (const float*)d_in[9];
  const float* Wgo    = (const float*)d_in[10];
  const float* Wpo    = (const float*)d_in[11];
  float* out = (float*)d_out;

  // workspace layout
  char* ws = (char*)d_ws;
  f16* wt   = (f16*)ws;                               // 6 * 16384 f16 (192 KB, in a 256 KB region)
  const size_t SZH = (size_t)CZ * LL;                 // 33.5M elements
  f16* a_ws = (f16*)(ws + (256 << 10));
  f16* b_ws = a_ws + SZH;
  f16* g_ws = b_ws + SZH;
  float* m_ws = (float*)(g_ws + SZH);

  k_wconv<<<6, 256, 0, stream>>>(Wga, Wpa, Wgb, Wpb, Wgo, Wpo, wt);
  k_proj<<<LL / 128, 256, 0, stream>>>(z, mask, lnin_w, lnin_b, wt, a_ws, b_ws, g_ws);
  k_einsum<<<dim3(L / 32, L / 32, CZ / 4), 128, 0, stream>>>(a_ws, b_ws, m_ws);
  k_final<<<LL / 128, 256, 0, stream>>>(m_ws, mask, lnout_w, lnout_b,
                                        wt + 5 * 16384, g_ws, out);
}
